// AttentionLayer_19782619365684
// MI455X (gfx1250) — matrix operand order repaired and checked by reading, not compiled, not run on hardware
//
#include <hip/hip_runtime.h>
#include <hip/hip_bf16.h>
#include <math.h>

// Shapes from the reference
#define B_SZ 32
#define F_SZ 2048
#define H_SZ 1024
#define P_SZ 512
#define M_SZ (B_SZ * F_SZ)   // 65536 GEMM rows

typedef __attribute__((ext_vector_type(16))) __bf16 v16bf;
typedef __attribute__((ext_vector_type(8)))  __bf16 v8bf;
typedef __attribute__((ext_vector_type(8)))  float  v8f;
typedef __attribute__((ext_vector_type(4)))  float  f32x4;

// ---------------------------------------------------------------------------
// Kernel 1: hp = h @ W1^T + b1   [32, 512] — tiny, plain dot products
// ---------------------------------------------------------------------------
__global__ void hp_kernel(const float* __restrict__ h,
                          const float* __restrict__ W1,
                          const float* __restrict__ b1,
                          float* __restrict__ hp) {
    int b = blockIdx.x;      // 0..31
    int q = threadIdx.x;     // 0..511
    const float* hr = h  + (size_t)b * H_SZ;
    const float* wr = W1 + (size_t)q * H_SZ;
    float s = b1[q];
    for (int k = 0; k < H_SZ; ++k) s = fmaf(hr[k], wr[k], s);
    hp[b * P_SZ + q] = s;
}

// ---------------------------------------------------------------------------
// Kernel 2: pack W2 (fp32 [q][p]) into bf16 WMMA B-fragment layout.
// B-matrix (K x N per step, K=32): lane = {n = l&15, khalf = l>>4},
// element e (0..15) holds K = ks*32 + (e>>3)*16 + khalf*8 + (e&7), column n = qt*16 + (l&15).
// Record per (qt, ks, lane) = 16 bf16 = 32 bytes, contiguous.
// ---------------------------------------------------------------------------
__global__ void pack_w2_kernel(const float* __restrict__ W2,
                               __bf16* __restrict__ w2p) {
    int idx = blockIdx.x * blockDim.x + threadIdx.x;   // 262144 total
    int e    = idx & 15;
    int l    = (idx >> 4) & 31;
    int ksqt = idx >> 9;            // qt*16 + ks
    int ks   = ksqt & 15;
    int qt   = ksqt >> 4;
    int n    = l & 15;
    int kh   = l >> 4;
    int k    = ks * 32 + (e >> 3) * 16 + kh * 8 + (e & 7);
    int q    = qt * 16 + n;
    w2p[idx] = (__bf16)W2[(size_t)q * P_SZ + k];
}

// ---------------------------------------------------------------------------
// Kernel 3: main fused GEMM:  vp = v @ W2^T (bf16 WMMA, f32 acc),
// then x = tanh(hp + vp + b2), e[b,f] = sum_q x*w3[q].
// Grid: 4096 blocks (one 16-row M-tile each) x 256 threads (8 waves).
// A tile (16 x 512) staged ONCE into LDS, bf16, in WMMA fragment-record
// layout (16 KB). Wave w handles N-cols [w*64, w*64+64): 4 acc tiles.
// B fragments are software-pipelined one K-step ahead (L2-resident, 512 KB).
// Cross-wave row reduction via LDS (no global atomics).
// ---------------------------------------------------------------------------
__global__ __launch_bounds__(256) void gemm_tanh_dot_kernel(
    const float*  __restrict__ vsrc,
    const float*  __restrict__ hp,
    const __bf16* __restrict__ w2p,
    const float*  __restrict__ b2,
    const float*  __restrict__ w3,
    float* __restrict__ e) {

    __shared__ __bf16 aLds[16 * 512];   // 16 KB: A-fragment records [ks][lane][16]
    __shared__ float  psum[8][16];      // per-wave partial row sums

    const int mtile = blockIdx.x;            // 0..4095
    const int wave  = threadIdx.x >> 5;      // 0..7
    const int lane  = threadIdx.x & 31;
    const int nl    = lane & 15;
    const int kh    = lane >> 4;
    const int mbase = mtile * 16;
    const int qbase = wave * 64;

    // ---- Stage A tile: global fp32 -> bf16 fragment records in LDS --------
    // Record (ks, l) element e holds v[mbase + (l&15)][ks*32 + (l>>4)*8 +
    // (e>>3)*16 + (e&7)]. A thread converts one aligned 8-float run per step.
    {
        const int t = threadIdx.x;
#pragma unroll
        for (int u = 0; u < 4; ++u) {
            int rid = t + u * 256;           // 0..1023 runs
            int row = rid >> 6;              // 0..15
            int k0  = (rid & 63) << 3;       // 0,8,...,504
            const f32x4* src =
                (const f32x4*)(vsrc + (size_t)(mbase + row) * P_SZ + k0);
            f32x4 x0 = src[0];
            f32x4 x1 = src[1];
            int ks  = k0 >> 5;
            int khs = (k0 >> 3) & 1;         // which lane half
            int rh  = (k0 >> 4) & 1;         // which record half (e<8 / e>=8)
            v8bf tmp;
            tmp[0] = (__bf16)x0[0]; tmp[1] = (__bf16)x0[1];
            tmp[2] = (__bf16)x0[2]; tmp[3] = (__bf16)x0[3];
            tmp[4] = (__bf16)x1[0]; tmp[5] = (__bf16)x1[1];
            tmp[6] = (__bf16)x1[2]; tmp[7] = (__bf16)x1[3];
            *(v8bf*)(aLds + (((ks * 32) + khs * 16 + row) << 4) + (rh << 3)) = tmp;
        }
    }
    __syncthreads();

    // ---- K loop: WMMA with B double-buffered one step ahead ---------------
    v8f acc[4];
#pragma unroll
    for (int nt = 0; nt < 4; ++nt)
#pragma unroll
        for (int r = 0; r < 8; ++r) acc[nt][r] = 0.0f;

    const v16bf* aRec = (const v16bf*)aLds;
    const v16bf* w2t  = (const v16bf*)w2p;
    const int qt0 = qbase >> 4;

    v16bf bfr[4];
#pragma unroll
    for (int nt = 0; nt < 4; ++nt)
        bfr[nt] = w2t[(((qt0 + nt) * 16) << 5) + lane];

#pragma unroll 4
    for (int ks = 0; ks < 16; ++ks) {
        v16bf af = aRec[ks * 32 + lane];
        int ksn = (ks + 1) & 15;             // branch-free pipeline index
        v16bf bn[4];
#pragma unroll
        for (int nt = 0; nt < 4; ++nt)
            bn[nt] = w2t[((((qt0 + nt) * 16) + ksn) << 5) + lane];
#pragma unroll
        for (int nt = 0; nt < 4; ++nt)
            acc[nt] = __builtin_amdgcn_wmma_f32_16x16x32_bf16(
                false, af, false, bfr[nt], (short)0, acc[nt], false, false);
#pragma unroll
        for (int nt = 0; nt < 4; ++nt) bfr[nt] = bn[nt];
    }

    // ---- Epilogue: tanh + dot(w3), block-wide row reduction ---------------
    // C/D layout: VGPR r, lanes 0-15 -> (M=r, N=lane); lanes 16-31 -> M=r+8.
    // Batch index is uniform over the M-tile (2048 % 16 == 0).
    const int bidx = mbase >> 11;            // g >> 11, F = 2048

    float partial[8];
#pragma unroll
    for (int r = 0; r < 8; ++r) partial[r] = 0.0f;

#pragma unroll
    for (int nt = 0; nt < 4; ++nt) {
        int q = qbase + nt * 16 + nl;
        float hq  = hp[bidx * P_SZ + q] + b2[q];
        float w3q = w3[q];
#pragma unroll
        for (int r = 0; r < 8; ++r) {
            float x = tanhf(acc[nt][r] + hq);
            partial[r] = fmaf(x, w3q, partial[r]);
        }
    }

    // Reduce across the 16 lanes of each half-wave (masks <16 keep halves
    // separate, matching the C-matrix row split).
#pragma unroll
    for (int r = 0; r < 8; ++r) {
        float s = partial[r];
        s += __shfl_xor(s, 1);
        s += __shfl_xor(s, 2);
        s += __shfl_xor(s, 4);
        s += __shfl_xor(s, 8);
        if (nl == 0) psum[wave][r + 8 * kh] = s;
    }
    __syncthreads();

    // Combine the 8 N-group waves; one store per output row.
    if (threadIdx.x < 16) {
        float s = 0.0f;
#pragma unroll
        for (int w = 0; w < 8; ++w) s += psum[w][threadIdx.x];
        e[mbase + threadIdx.x] = s;
    }
}

// ---------------------------------------------------------------------------
// Kernel 4: scrambled softmax.
// e_scrambled[i][j] = e[(i*F+j) % B][(i*F+j) / B] = e[j&31][i*64 + (j>>5)]
// (F divisible by B). One block per output row.
// ---------------------------------------------------------------------------
__global__ __launch_bounds__(256) void softmax_scramble_kernel(
    const float* __restrict__ e, float* __restrict__ out) {
    __shared__ float sred[256];
    const int i = blockIdx.x;      // 0..31
    const int t = threadIdx.x;     // 0..255

    float vals[8];
    float mx = -INFINITY;
#pragma unroll
    for (int u = 0; u < 8; ++u) {
        int j = t + u * 256;                       // 0..2047
        int src = (j & 31) * F_SZ + i * 64 + (j >> 5);
        vals[u] = e[src];
        mx = fmaxf(mx, vals[u]);
    }
    sred[t] = mx;
    __syncthreads();
    for (int s = 128; s > 0; s >>= 1) {
        if (t < s) sred[t] = fmaxf(sred[t], sred[t + s]);
        __syncthreads();
    }
    mx = sred[0];
    __syncthreads();

    float sum = 0.0f;
#pragma unroll
    for (int u = 0; u < 8; ++u) {
        vals[u] = __expf(vals[u] - mx);
        sum += vals[u];
    }
    sred[t] = sum;
    __syncthreads();
    for (int s = 128; s > 0; s >>= 1) {
        if (t < s) sred[t] += sred[t + s];
        __syncthreads();
    }
    float inv = 1.0f / sred[0];
#pragma unroll
    for (int u = 0; u < 8; ++u) {
        int j = t + u * 256;
        out[i * F_SZ + j] = vals[u] * inv;
    }
}

// ---------------------------------------------------------------------------
// Host-side launcher
// ---------------------------------------------------------------------------
extern "C" void kernel_launch(void* const* d_in, const int* in_sizes, int n_in,
                              void* d_out, int out_size, void* d_ws, size_t ws_size,
                              hipStream_t stream) {
    (void)in_sizes; (void)n_in; (void)out_size; (void)ws_size;

    const float* h  = (const float*)d_in[0];   // [32,1024]
    const float* v  = (const float*)d_in[1];   // [32,2048,512]
    const float* W1 = (const float*)d_in[2];   // [512,1024]
    const float* b1 = (const float*)d_in[3];   // [512]
    const float* W2 = (const float*)d_in[4];   // [512,512]
    const float* b2 = (const float*)d_in[5];   // [512]
    const float* w3 = (const float*)d_in[6];   // [512]
    float* out = (float*)d_out;                // [32,2048]

    char* ws = (char*)d_ws;
    float*  hp  = (float*)(ws);                        // 32*512*4   = 65536 B
    __bf16* w2p = (__bf16*)(ws + 65536);               // 512*512*2  = 524288 B
    float*  e   = (float*)(ws + 65536 + 524288);       // 65536*4    = 262144 B

    hp_kernel<<<B_SZ, P_SZ, 0, stream>>>(h, W1, b1, hp);
    pack_w2_kernel<<<(P_SZ * P_SZ) / 256, 256, 0, stream>>>(W2, w2p);
    gemm_tanh_dot_kernel<<<M_SZ / 16, 256, 0, stream>>>(v, hp, w2p, b2, w3, e);
    softmax_scramble_kernel<<<B_SZ, 256, 0, stream>>>(e, out);
}
